// InvariantParticleAttention_40742059770401
// MI455X (gfx1250) — compile-verified, matrix operand order checked by reading, not run on hardware
//
#include <hip/hip_runtime.h>
#include <hip/hip_bf16.h>

typedef __attribute__((ext_vector_type(16))) __bf16 v16bf;
typedef __attribute__((ext_vector_type(8)))  float  v8f;
typedef int v4i_vec __attribute__((vector_size(4 * sizeof(int))));

#define PITCH 72
#define LOG2E 1.44269504088896340736f

#if __has_builtin(__builtin_amdgcn_global_load_async_to_lds_b128)
#define HAVE_ASYNC_LDS 1
#else
#define HAVE_ASYNC_LDS 0
#endif

#if HAVE_ASYNC_LDS
__device__ __forceinline__ void async_cp16(const __bf16* g, __bf16* l) {
    __builtin_amdgcn_global_load_async_to_lds_b128(
        (__attribute__((address_space(1))) v4i_vec*)g,
        (__attribute__((address_space(3))) v4i_vec*)l, 0, 0);
}
__device__ __forceinline__ void async_wait0() {
#if __has_builtin(__builtin_amdgcn_s_wait_asynccnt)
    __builtin_amdgcn_s_wait_asynccnt(0);
#else
    asm volatile("s_wait_asynccnt 0x0" ::: "memory");
#endif
}
#endif

union FragU {
    uint4 u[2];
    v16bf v;
};
union Pack16 {
    uint4  u[2];
    __bf16 h[16];
};

// Load a 16x32 bf16 fragment (A layout; B mirrored with N in lanes)
// from a row-major LDS tile.  Two 16-byte LDS loads per lane.
__device__ __forceinline__ v16bf ldfrag(const __bf16* tile, int rowbase, int colbase) {
    const int l   = threadIdx.x & 31;
    const int row = rowbase + (l & 15);
    const int cb  = colbase + ((l >> 4) << 3);
    const __bf16* p = tile + row * PITCH + cb;
    FragU f;
    f.u[0] = *(const uint4*)(p);
    f.u[1] = *(const uint4*)(p + 16);
    return f.v;
}

// ---- 16-lane butterfly reductions entirely in the VALU via DPP16 ----
// quad_perm(1,0,3,2)=0xB1, quad_perm(2,3,0,1)=0x4E,
// row_half_mirror=0x141, row_mirror=0x140.  DPP rows are 16 lanes, so the
// two wave32 halves reduce independently (exactly the C-matrix granularity).
template <int CTRL>
__device__ __forceinline__ float dpp_bcast(float x) {
    return __int_as_float(
        __builtin_amdgcn_update_dpp(0, __float_as_int(x), CTRL, 0xF, 0xF, true));
}
__device__ __forceinline__ float hmax16(float x) {
    x = fmaxf(x, dpp_bcast<0xB1>(x));
    x = fmaxf(x, dpp_bcast<0x4E>(x));
    x = fmaxf(x, dpp_bcast<0x141>(x));
    x = fmaxf(x, dpp_bcast<0x140>(x));
    return x;
}
__device__ __forceinline__ float hsum16(float x) {
    x += dpp_bcast<0xB1>(x);
    x += dpp_bcast<0x4E>(x);
    x += dpp_bcast<0x141>(x);
    x += dpp_bcast<0x140>(x);
    return x;
}

// ---------------------------------------------------------------------------
// Pre-pass: apply per-token frames to q/k/v, fold 1/sqrt(D) into q, emit bf16.
//   Linv[i][j]       = s_i * s_j * L[j][i]   (for q, v)
//   Linv_lower[i][j] =       s_j * L[j][i]   (for k)
// ---------------------------------------------------------------------------
__global__ __launch_bounds__(256) void ipa_prep(
    const float* __restrict__ q, const float* __restrict__ k,
    const float* __restrict__ v, const float* __restrict__ lf,
    __bf16* __restrict__ qo, __bf16* __restrict__ ko, __bf16* __restrict__ vo,
    int H, int N, size_t total)
{
    size_t idx = (size_t)blockIdx.x * blockDim.x + threadIdx.x;
    if (idx >= total) return;
    const int    d = (int)(idx & 63);
    const size_t t = idx >> 6;                 // flat (b,h,n)
    const int    nn = (int)(t % (size_t)N);
    const int    b  = (int)(t / ((size_t)N * H));

    if (d < 16) {
        qo[idx] = (__bf16)(q[idx] * 0.125f);
        ko[idx] = (__bf16)(k[idx]);
        vo[idx] = (__bf16)(v[idx]);
        return;
    }
    const int c = (d - 16) >> 2;
    const int i = (d - 16) & 3;
    const float sgn[4] = {1.f, -1.f, -1.f, -1.f};
    const float* lp = lf + ((size_t)b * N + nn) * 16;   // L row-major [i][j]
    const size_t base = (t << 6) + 16 + ((size_t)c << 2);
    const float si = sgn[i];
    float aq = 0.f, ak = 0.f, av = 0.f;
#pragma unroll
    for (int j = 0; j < 4; ++j) {
        const float Lji = lp[j * 4 + i];
        const float cj  = sgn[j] * Lji;  // k coefficient
        const float cq  = si * cj;       // q, v coefficient
        aq += cq * q[base + j];
        av += cq * v[base + j];
        ak += cj * k[base + j];
    }
    qo[idx] = (__bf16)(aq * 0.125f);
    ko[idx] = (__bf16)(ak);
    vo[idx] = (__bf16)(av);
}

// ---------------------------------------------------------------------------
// Flash attention: one block = (b, h, 128 q rows); 8 waves, 16 q rows each.
// K/V double-buffered in LDS (second buffer reuses the dead Q staging tile),
// one barrier per K-tile, next tile loaded (async-to-LDS for K, VGPR-staged
// for the transposed V) before computing the current tile.
// ---------------------------------------------------------------------------
__global__ __launch_bounds__(256) void ipa_flash(
    const __bf16* __restrict__ qg, const __bf16* __restrict__ kg,
    const __bf16* __restrict__ vg, const float* __restrict__ lf,
    float* __restrict__ out, int B, int H, int N)
{
    // region layout (all bf16, pitch 72):
    //   [0      .. 128*P) : Q staging, reused as buffer-1 (K1 | Vt1)
    //   [128*P  .. 192*P) : K0
    //   [192*P  .. 256*P) : Vt0 (transposed [d][token])
    //   [256*P  .. 384*P) : P
    __shared__ __bf16 smem[384 * PITCH];
    __bf16* sQ   = smem;
    __bf16* sK1  = smem;
    __bf16* sVt1 = smem + 64 * PITCH;
    __bf16* sK0  = smem + 128 * PITCH;
    __bf16* sVt0 = smem + 192 * PITCH;
    __bf16* sP   = smem + 256 * PITCH;

    const int nqb = N >> 7;
    const int qb  = blockIdx.x % nqb;
    const int bh  = blockIdx.x / nqb;
    const int b   = bh / H;
    const int q0  = qb << 7;

    const int tid  = threadIdx.x;
    const int wave = tid >> 5;
    const int lane = tid & 31;
    const int half = lane >> 4;
    const int n16  = lane & 15;

    // per-thread K/V staging coordinates (64 tokens x 64 dims per tile)
    const int ltok = tid >> 2;
    const int ld0  = (tid & 3) << 4;

    const size_t headoff = (size_t)bh * N * 64;

    // ---- issue tile-0 K/V loads, then stage Q ----
    Pack16 pv0;
#if !HAVE_ASYNC_LDS
    Pack16 pk0;
#endif
    {
        const size_t o0 = headoff + (size_t)ltok * 64 + ld0;
#if HAVE_ASYNC_LDS
        async_cp16(kg + o0,     sK0 + ltok * PITCH + ld0);
        async_cp16(kg + o0 + 8, sK0 + ltok * PITCH + ld0 + 8);
#else
        const uint4* ks = (const uint4*)(kg + o0);
        pk0.u[0] = ks[0]; pk0.u[1] = ks[1];
#endif
        const uint4* vs = (const uint4*)(vg + o0);
        pv0.u[0] = vs[0]; pv0.u[1] = vs[1];
    }
    {
        const int row = tid >> 1;
        const int d0  = (tid & 1) << 5;
        const uint4* src = (const uint4*)(qg + headoff + (size_t)(q0 + row) * 64 + d0);
        uint4* dst = (uint4*)(sQ + row * PITCH + d0);
        dst[0] = src[0]; dst[1] = src[1]; dst[2] = src[2]; dst[3] = src[3];
    }
    __syncthreads();

    // persistent Q A-fragments (16x32 each, D split 0..31 / 32..63)
    const v16bf qf0 = ldfrag(sQ, wave * 16, 0);
    const v16bf qf1 = ldfrag(sQ, wave * 16, 32);

    // finish staging tile 0
    {
#if !HAVE_ASYNC_LDS
        uint4* kd = (uint4*)(sK0 + ltok * PITCH + ld0);
        kd[0] = pk0.u[0]; kd[1] = pk0.u[1];
#endif
#pragma unroll
        for (int i = 0; i < 16; ++i)
            sVt0[(ld0 + i) * PITCH + ltok] = pv0.h[i];
#if HAVE_ASYNC_LDS
        async_wait0();
#endif
    }
    __syncthreads();   // buffer 0 visible; all waves past Q reads -> buffer 1 writable

    v8f O[4];
    float mrow[8], lrow[8];
#pragma unroll
    for (int c = 0; c < 4; ++c) O[c] = (v8f){0.f,0.f,0.f,0.f,0.f,0.f,0.f,0.f};
#pragma unroll
    for (int r = 0; r < 8; ++r) { mrow[r] = -3.0e38f; lrow[r] = 0.f; }

    const int nkt = N >> 6;
    int cur = 0;
    for (int kt = 0; kt < nkt; ++kt) {
        __bf16* cK  = cur ? sK1  : sK0;
        __bf16* cVt = cur ? sVt1 : sVt0;
        __bf16* nK  = cur ? sK0  : sK1;
        __bf16* nVt = cur ? sVt0 : sVt1;
        const bool haveNext = (kt + 1 < nkt);

        // ---- issue next tile's loads before computing current tile ----
        Pack16 pv;
#if !HAVE_ASYNC_LDS
        Pack16 pk;
#endif
        if (haveNext) {
            const size_t no = headoff + (size_t)(((kt + 1) << 6) + ltok) * 64 + ld0;
#if HAVE_ASYNC_LDS
            async_cp16(kg + no,     nK + ltok * PITCH + ld0);
            async_cp16(kg + no + 8, nK + ltok * PITCH + ld0 + 8);
#else
            const uint4* ks = (const uint4*)(kg + no);
            pk.u[0] = ks[0]; pk.u[1] = ks[1];
#endif
            const uint4* vs = (const uint4*)(vg + no);
            pv.u[0] = vs[0]; pv.u[1] = vs[1];
            if (kt + 2 < nkt) {
                const size_t po = headoff + (size_t)(((kt + 2) << 6) + ltok) * 64 + ld0;
                __builtin_prefetch(kg + po, 0, 1);
                __builtin_prefetch(vg + po, 0, 1);
            }
        }

        // ---- scores: S(16x64) = Q(16x64) * K^T, 4 subtiles x 2 WMMA ----
        v8f S[4];
#pragma unroll
        for (int s = 0; s < 4; ++s) {
            v8f c = (v8f){0.f,0.f,0.f,0.f,0.f,0.f,0.f,0.f};
            c = __builtin_amdgcn_wmma_f32_16x16x32_bf16(
                    false, qf0, false, ldfrag(cK, s * 16, 0),  (short)0, c, false, false);
            c = __builtin_amdgcn_wmma_f32_16x16x32_bf16(
                    false, qf1, false, ldfrag(cK, s * 16, 32), (short)0, c, false, false);
            S[s] = c;
        }

        // ---- online softmax (row = r + 8*half in C layout) ----
#pragma unroll
        for (int r = 0; r < 8; ++r) {
            float mx = fmaxf(fmaxf(S[0][r], S[1][r]), fmaxf(S[2][r], S[3][r]));
            mx = hmax16(mx);
            const float mn    = fmaxf(mrow[r], mx);
            const float alpha = exp2f((mrow[r] - mn) * LOG2E);
            float rs = 0.f;
            __bf16* prow = sP + (wave * 16 + r + 8 * half) * PITCH + n16;
#pragma unroll
            for (int s = 0; s < 4; ++s) {
                const float p = exp2f((S[s][r] - mn) * LOG2E);
                rs += p;
                prow[s * 16] = (__bf16)p;
            }
            rs = hsum16(rs);
            lrow[r] = lrow[r] * alpha + rs;
            mrow[r] = mn;
#pragma unroll
            for (int c = 0; c < 4; ++c) O[c][r] *= alpha;
        }

        // ---- O += P(16x64) * V(64x64): per-wave-private P (same-wave DS order) ----
#pragma unroll
        for (int kh = 0; kh < 2; ++kh) {
            const v16bf pa = ldfrag(sP, wave * 16, kh * 32);
#pragma unroll
            for (int c = 0; c < 4; ++c) {
                O[c] = __builtin_amdgcn_wmma_f32_16x16x32_bf16(
                           false, pa, false, ldfrag(cVt, c * 16, kh * 32),
                           (short)0, O[c], false, false);
            }
        }

        // ---- commit next tile into the other buffer ----
        if (haveNext) {
#if !HAVE_ASYNC_LDS
            uint4* kd = (uint4*)(nK + ltok * PITCH + ld0);
            kd[0] = pk.u[0]; kd[1] = pk.u[1];
#endif
#pragma unroll
            for (int i = 0; i < 16; ++i)
                nVt[(ld0 + i) * PITCH + ltok] = pv.h[i];
#if HAVE_ASYNC_LDS
            async_wait0();
#endif
        }
        __syncthreads();
        cur ^= 1;
    }

    // ---- epilogue: O/l, then out = L * vec per 4-vector block ----
    const float* lfb = lf + (size_t)b * N * 16;
    const int i4 = (n16 & 3) * 4;
    const int nb = n16 & ~3;
#pragma unroll
    for (int r = 0; r < 8; ++r) {
        const int   token = q0 + wave * 16 + r + 8 * half;
        const float invl  = 1.0f / lrow[r];
        const float4 Lr   = *(const float4*)(lfb + (size_t)token * 16 + i4);
        float* orow = out + headoff + (size_t)token * 64;
        orow[n16] = O[0][r] * invl;                // scalars d=0..15
#pragma unroll
        for (int c = 1; c < 4; ++c) {
            const float val = O[c][r] * invl;
            const float a0 = __shfl(val, half * 16 + nb + 0, 32);
            const float a1 = __shfl(val, half * 16 + nb + 1, 32);
            const float a2 = __shfl(val, half * 16 + nb + 2, 32);
            const float a3 = __shfl(val, half * 16 + nb + 3, 32);
            orow[c * 16 + n16] = Lr.x * a0 + Lr.y * a1 + Lr.z * a2 + Lr.w * a3;
        }
    }
}

extern "C" void kernel_launch(void* const* d_in, const int* in_sizes, int n_in,
                              void* d_out, int out_size, void* d_ws, size_t ws_size,
                              hipStream_t stream) {
    const float* q  = (const float*)d_in[0];
    const float* k  = (const float*)d_in[1];
    const float* v  = (const float*)d_in[2];
    const float* lf = (const float*)d_in[3];

    const int H = 8, N = 2048, D = 64;
    const int B = in_sizes[0] / (H * N * D);
    const size_t per = (size_t)B * H * N * D;

    __bf16* qo = (__bf16*)d_ws;
    __bf16* ko = qo + per;
    __bf16* vo = ko + per;

    const int threads = 256;
    const int blocks  = (int)((per + threads - 1) / threads);
    ipa_prep<<<blocks, threads, 0, stream>>>(q, k, v, lf, qo, ko, vo, H, N, per);

    const int grid = B * H * (N / 128);
    ipa_flash<<<grid, 256, 0, stream>>>(qo, ko, vo, lf, (float*)d_out, B, H, N);
}
